// GenericLinear_17772574671348
// MI455X (gfx1250) — matrix operand order repaired and checked by reading, not compiled, not run on hardware
//
#include <hip/hip_runtime.h>

// ---------------------------------------------------------------------------
// GA (Cl(3,0)) linear layer for MI455X / gfx1250.
// out[k*512+m, u] = bias[k*1024+u]
//                 + sum_i S[k][i] * sum_c x[i*512+m, c] * W[c, J[k][i]*1024+u]
// Cayley einsum fused into the GEMM; bf16 hi/lo split (3 WMMAs / K32 chunk).
// Staging uses gfx1250 GLOBAL_LOAD_ASYNC_TO_LDS_B128 (ASYNCcnt) with
// double-buffered LDS; Cayley sign XORed into B fragments (co-executes
// with WMMA since 16-bit XDL ops are TRANS-tracked).
// ---------------------------------------------------------------------------

typedef __attribute__((ext_vector_type(16))) __bf16       v16bf;
typedef __attribute__((ext_vector_type(8)))  float        v8f;
typedef __attribute__((ext_vector_type(4)))  unsigned int v4u;
typedef __attribute__((ext_vector_type(4)))  float        v4f;

union Frag { v16bf v; v4u q[2]; };

// c_JS[k][i] = J[k][i] | (S[k][i]<0 ? 8 : 0): blade pairing + sign for Cl(3,0)
__constant__ int c_JS[8][8] = {
  {0, 1, 2, 3, 12, 13, 14, 15},   // k=0 (scalar)
  {1, 0, 12, 13, 2, 11, 15, 14},  // k=1 (e1)
  {2, 4, 0, 14, 9, 15, 3, 13},    // k=2 (e2)
  {3, 5, 6, 0, 15, 9, 10, 12},    // k=3 (e3)
  {4, 2, 9, 7, 0, 14, 5, 3},      // k=4 (e12)
  {5, 3, 15, 9, 6, 0, 12, 10},    // k=5 (e13)
  {6, 7, 3, 10, 13, 4, 0, 1},     // k=6 (e23)
  {7, 6, 13, 4, 3, 10, 1, 0},     // k=7 (e123)
};

__device__ __forceinline__ unsigned short f2bf(float f) {
  unsigned u = __float_as_uint(f);
  u += 0x7FFFu + ((u >> 16) & 1u);          // round-to-nearest-even
  return (unsigned short)(u >> 16);
}
__device__ __forceinline__ float bf2f(unsigned short h) {
  return __uint_as_float(((unsigned)h) << 16);
}

// ---- preprocessing: x (4096x1024 f32) -> bf16 hi/lo, same layout -----------
__global__ void convert_x_kernel(const float* __restrict__ x,
                                 unsigned short* __restrict__ xh,
                                 unsigned short* __restrict__ xl) {
  int idx = (blockIdx.x * 256 + threadIdx.x) * 4;
  v4f v = *(const v4f*)(x + idx);
#pragma unroll
  for (int e = 0; e < 4; ++e) {
    float f = v[e];
    unsigned short h = f2bf(f);
    xh[idx + e] = h;
    xl[idx + e] = f2bf(f - bf2f(h));
  }
}

// ---- preprocessing: W (1024x8192 f32) -> Wt (8192x1024) bf16 hi/lo ---------
__global__ void convert_w_kernel(const float* __restrict__ W,
                                 unsigned short* __restrict__ wh,
                                 unsigned short* __restrict__ wl) {
  __shared__ float tile[32][33];
  int c0 = blockIdx.x * 32;   // W column  -> Wt row
  int r0 = blockIdx.y * 32;   // W row (K) -> Wt column
  int tx = threadIdx.x, ty = threadIdx.y;
#pragma unroll
  for (int q = 0; q < 4; ++q)
    tile[ty + q * 8][tx] = W[(size_t)(r0 + ty + q * 8) * 8192 + c0 + tx];
  __syncthreads();
#pragma unroll
  for (int q = 0; q < 4; ++q) {
    float f = tile[tx][ty + q * 8];
    unsigned short h = f2bf(f);
    size_t o = (size_t)(c0 + ty + q * 8) * 1024 + r0 + tx;
    wh[o] = h;
    wl[o] = f2bf(f - bf2f(h));
  }
}

// ---- gfx1250 async memory->LDS staging (ASYNCcnt path, ISA 15.18.3) --------
// GVS mode: lds-dest VGPR, 32-bit byte-offset VGPR, 64-bit SGPR base.
__device__ __forceinline__ void async_ld_b128(unsigned ldsaddr, unsigned voff,
                                              unsigned long long gbase) {
  asm volatile("global_load_async_to_lds_b128 %0, %1, %2"
               :
               : "v"(ldsaddr), "v"(voff), "s"(gbase)
               : "memory");
}
__device__ __forceinline__ void wait_async_le12() {
  asm volatile("s_wait_asynccnt 0xc" ::: "memory");
}
__device__ __forceinline__ void wait_async_0() {
  asm volatile("s_wait_asynccnt 0x0" ::: "memory");
}

// ---- LDS fragment loads, layouts per CDNA5 ISA 7.12.2 (wave32) -------------
// LDS rows padded to 40 halfwords (80B): 16B-aligned, bank-rotating.
__device__ __forceinline__ v16bf frag_a(const unsigned short* base, int m, int lane) {
  int row = m + (lane & 15);
  int hb = lane >> 4;
  const char* p = (const char*)(base + row * 40) + hb * 16;
  Frag f;
  f.q[0] = *(const v4u*)p;          // K = hb*8 + 0..7
  f.q[1] = *(const v4u*)(p + 32);   // K = 16 + hb*8 + 0..7
  return f.v;
}
// B fragment with Cayley sign folded in: XOR of bf16 sign bits (sx = 0 or
// 0x80008000). These VALU ops co-execute with the TRANS-tracked WMMAs.
__device__ __forceinline__ v16bf frag_b_sgn(const unsigned short* base, int n,
                                            int lane, unsigned sx) {
  int row = n + (lane & 15);
  int hb = lane >> 4;
  const char* p = (const char*)(base + row * 40) + hb * 16;
  Frag f;
  f.q[0] = *(const v4u*)p;          // K = hb*8 + 0..7
  f.q[1] = *(const v4u*)(p + 32);   // K = 16 + hb*8 + 0..7
  f.q[0].x ^= sx; f.q[0].y ^= sx; f.q[0].z ^= sx; f.q[0].w ^= sx;
  f.q[1].x ^= sx; f.q[1].y ^= sx; f.q[1].z ^= sx; f.q[1].w ^= sx;
  return f.v;
}

#define WMMA_BF16(a, b, c) \
  __builtin_amdgcn_wmma_f32_16x16x32_bf16(false, (a), false, (b), (short)0, (c), false, false)

// ---- fused GA-GEMM ---------------------------------------------------------
// grid = (64, 8): x = 4 M-blocks * 16 N-blocks, y = output blade k.
// Block tile 128x64, 4 waves, wave tile 64x32 (4x2 of 16x16).
// Effective K loop: 8 blades * 1024 = 256 chunks of K=32, double-buffered LDS.
__launch_bounds__(128)
__global__ void ga_gemm_kernel(const unsigned short* __restrict__ xs,   // xh | xl (contiguous)
                               const unsigned short* __restrict__ wt,   // wh | wl (contiguous)
                               const float* __restrict__ bias,
                               float* __restrict__ out) {
  // [dbuf][hi/lo][row][40]
  __shared__ __align__(16) unsigned short As[2][2][128][40];   // 40,960 B
  __shared__ __align__(16) unsigned short Bs[2][2][64][40];    // 20,480 B

  const int tid  = threadIdx.x;
  const int lane = tid & 31;
  const int wv   = tid >> 5;
  const int wm   = (wv >> 1) * 64;           // wave M offset in tile
  const int wn   = (wv & 1) * 32;            // wave N offset in tile
  const int kb   = blockIdx.y;               // output blade
  const int bm   = (blockIdx.x & 3) * 128;   // M offset inside blade (0..511)
  const int bn   = (blockIdx.x >> 2) * 64;   // N offset (0..1023)

  const unsigned long long xsBase = (unsigned long long)(uintptr_t)xs;
  const unsigned long long wtBase = (unsigned long long)(uintptr_t)wt;

  // Per-thread async staging decode. A: 1024 16B chunks, 8/thread; B: 512, 4/thread.
  unsigned aOff[8], aLds[8];     // global byte offset (sans chunk), LDS addr (dbuf 0)
  unsigned bOff[4], bLds[4];
#pragma unroll
  for (int q = 0; q < 8; ++q) {
    int li = q * 128 + tid;                 // 0..1023
    int hbuf = li >> 9;                     // hi/lo
    int row  = (li >> 2) & 127;
    int seg  = li & 3;
    aOff[q] = (unsigned)hbuf * 8388608u     // 4096*1024*2 : xl after xh
            + (unsigned)(bm + row) * 2048u + (unsigned)seg * 16u;
    aLds[q] = (unsigned)(uintptr_t)&As[0][hbuf][row][seg * 8];
  }
#pragma unroll
  for (int q = 0; q < 4; ++q) {
    int li = q * 128 + tid;                 // 0..511
    int hbuf = li >> 8;
    int row  = (li >> 2) & 63;
    int seg  = li & 3;
    bOff[q] = (unsigned)hbuf * 16777216u    // 8192*1024*2 : wl after wh
            + (unsigned)(bn + row) * 2048u + (unsigned)seg * 16u;
    bLds[q] = (unsigned)(uintptr_t)&Bs[0][hbuf][row][seg * 8];
  }

  // issue one chunk's 12 async loads into double-buffer db
  auto issue = [&](int cn, int db) {
    const int in  = cn >> 5;
    const int kc2 = ((cn & 31) << 5) * 2;           // byte offset along K
    const unsigned aAdd = (unsigned)in * 1048576u + (unsigned)kc2;  // in*512*2048
    const unsigned dbA  = (unsigned)db * 20480u;
#pragma unroll
    for (int q = 0; q < 8; ++q)
      async_ld_b128(aLds[q] + dbA, aOff[q] + aAdd, xsBase);
    const int jb = c_JS[kb][in] & 7;
    const unsigned bAdd = (unsigned)jb * 2097152u + (unsigned)kc2;  // jb*1024*2048
    const unsigned dbB  = (unsigned)db * 10240u;
#pragma unroll
    for (int q = 0; q < 4; ++q)
      async_ld_b128(bLds[q] + dbB, bOff[q] + bAdd, wtBase);
  };

  v8f acc[4][2];
#pragma unroll
  for (int tm = 0; tm < 4; ++tm)
#pragma unroll
    for (int tn = 0; tn < 2; ++tn) {
      v8f z = {0.f, 0.f, 0.f, 0.f, 0.f, 0.f, 0.f, 0.f};
      acc[tm][tn] = z;
    }

  issue(0, 0);
  issue(1, 1);

#pragma unroll 1
  for (int c = 0; c < 256; ++c) {
    const int i  = c >> 5;        // input blade segment
    const int db = c & 1;
    const unsigned sx = (c_JS[kb][i] & 8) ? 0x80008000u : 0u;  // Cayley sign

    // in-order async completion: <=12 outstanding => chunk c's batch landed
    if (c < 255) wait_async_le12(); else wait_async_0();
    __syncthreads();              // all waves' async data visible in LDS

    v16bf ah[4], al[4], bh[2], bl[2];
#pragma unroll
    for (int tm = 0; tm < 4; ++tm) {
      ah[tm] = frag_a(&As[db][0][0][0], wm + tm * 16, lane);
      al[tm] = frag_a(&As[db][1][0][0], wm + tm * 16, lane);
    }
#pragma unroll
    for (int tn = 0; tn < 2; ++tn) {
      bh[tn] = frag_b_sgn(&Bs[db][0][0][0], wn + tn * 16, lane, sx);
      bl[tn] = frag_b_sgn(&Bs[db][1][0][0], wn + tn * 16, lane, sx);
    }

    // bf16x3: hi*hi + hi*lo + lo*hi  (lo*lo below fp32 noise floor)
#pragma unroll
    for (int tm = 0; tm < 4; ++tm)
#pragma unroll
      for (int tn = 0; tn < 2; ++tn) {
        acc[tm][tn] = WMMA_BF16(ah[tm], bh[tn], acc[tm][tn]);
        acc[tm][tn] = WMMA_BF16(ah[tm], bl[tn], acc[tm][tn]);
        acc[tm][tn] = WMMA_BF16(al[tm], bh[tn], acc[tm][tn]);
      }

    __syncthreads();              // buffer db fully consumed by all waves
    if (c + 2 < 256) issue(c + 2, db);
  }

  // epilogue: C layout (ISA 7.12.2): VGPR r -> row r + 8*(lane>=16), col=lane&15
  const int colBase = bn + wn;
  const int rowBase = kb * 512 + bm + wm;
#pragma unroll
  for (int tm = 0; tm < 4; ++tm)
#pragma unroll
    for (int tn = 0; tn < 2; ++tn) {
      int col = colBase + tn * 16 + (lane & 15);
      float bv = bias[kb * 1024 + col];
      int r0 = rowBase + tm * 16 + (lane >> 4) * 8;
      v8f v = acc[tm][tn];
#pragma unroll
      for (int r = 0; r < 8; ++r)
        out[(size_t)(r0 + r) * 1024 + col] = v[r] + bv;
    }
}

// ---------------------------------------------------------------------------
extern "C" void kernel_launch(void* const* d_in, const int* in_sizes, int n_in,
                              void* d_out, int out_size, void* d_ws, size_t ws_size,
                              hipStream_t stream) {
  const float* x    = (const float*)d_in[0];  // (4096, 1024)
  const float* W    = (const float*)d_in[1];  // (1024, 8192)
  const float* bias = (const float*)d_in[2];  // (1, 8192)
  float* out        = (float*)d_out;          // (4096, 1024)

  // workspace: xh 8MB | xl 8MB | wh 16MB | wl 16MB = 48MB (hi/lo contiguous)
  char* ws = (char*)d_ws;
  unsigned short* xh = (unsigned short*)(ws);
  unsigned short* xl = (unsigned short*)(ws + 8ull * 1024 * 1024);
  unsigned short* wh = (unsigned short*)(ws + 16ull * 1024 * 1024);
  unsigned short* wl = (unsigned short*)(ws + 32ull * 1024 * 1024);

  convert_x_kernel<<<4096, 256, 0, stream>>>(x, xh, xl);
  convert_w_kernel<<<dim3(256, 32), dim3(32, 8), 0, stream>>>(W, wh, wl);
  ga_gemm_kernel<<<dim3(64, 8), 128, 0, stream>>>(xh, wh, bias, out);

  (void)xl; (void)wl;  // addressed via contiguous offsets from xh/wh
  (void)in_sizes; (void)n_in; (void)out_size; (void)ws_size;
}
